// DSDMSR_x8_69380901699735
// MI455X (gfx1250) — matrix operand order repaired and checked by reading, not compiled, not run on hardware
//
#include <hip/hip_runtime.h>
#include <hip/hip_bf16.h>

// DSDMSR x8 (SRCNN pyramid) for gfx1250.
// conv1/conv2 on the matrix pipe via v_wmma_f32_16x16x32_f16, f32 accumulate.
// h1 is channel-last f16 with a 2-pixel zero halo so conv2's inner loop is
// branchless: per 5x5 tap -> 1 addr calc, 4 aligned 32B B-loads (immediate
// offsets), 2 A-loads (pointer bump), 4 WMMAs. conv1 builds im2col B-fragments
// once per workgroup (wave w builds n-tile w) from a zero-padded LDS tile with
// compile-time tap offsets, then shares them via LDS as vector loads.
// Workspace requirement: ~55 MB.

typedef __attribute__((ext_vector_type(16))) _Float16 v16h;
typedef __attribute__((ext_vector_type(8)))  float    v8f;

// ---------------------------------------------------------------------------
// Weight packing into WMMA A-fragment lane order (16-bit A 16x32, ISA 7.12.2):
// element e of lane L: k = (L>=16?16:0) + e (contiguous k, not the ISA table order; B uses the same)
// ---------------------------------------------------------------------------
__global__ void pack_w1_kernel(const float* __restrict__ W1, _Float16* __restrict__ pW1, int total) {
    int idx = blockIdx.x * 256 + threadIdx.x;
    if (idx >= total) return;
    int e    = idx & 15;
    int lane = (idx >> 4) & 31;
    int rest = idx >> 9;                 // u*12 + mt*3 + ks
    int ks = rest % 3;
    int mt = (rest / 3) % 4;
    int u  = rest / 12;
    int m = lane & 15, half = lane >> 4;
    int v = e >> 1, r = e & 1;
    int k = ks * 32 + half * 16 + e;  // tap 0..95
    int ch = mt * 16 + m;
    float val = 0.f;
    if (k < 81) val = W1[((size_t)u * 64 + ch) * 81 + k];    // W1: (13,64,1,9,9)
    pW1[idx] = (_Float16)val;
}

__global__ void pack_w2_kernel(const float* __restrict__ W2, _Float16* __restrict__ pW2, int total) {
    int idx = blockIdx.x * 256 + threadIdx.x;
    if (idx >= total) return;
    int e    = idx & 15;
    int lane = (idx >> 4) & 31;
    int rest = idx >> 9;                 // u*100 + mt*50 + s
    int s  = rest % 50;
    int mt = (rest / 50) % 2;
    int u  = rest / 100;
    int tap = s >> 1, chunk = s & 1;     // K order: tap-major, 2 chunks of 32 channels
    int m = lane & 15, half = lane >> 4;
    int v = e >> 1, r = e & 1;
    int kl = half * 16 + e;
    int c   = chunk * 32 + kl;
    int och = mt * 16 + m;
    float val = W2[(((size_t)u * 32 + och) * 64 + c) * 25 + tap];  // W2: (13,32,64,5,5)
    pW2[idx] = (_Float16)val;
}

// ---------------------------------------------------------------------------
// Zero the 2-pixel halo border of the padded h1 feature map (per geometry).
// h1 layout: [(H+4) x (W+4)] pixels x 64 channels, channel-last f16.
// ---------------------------------------------------------------------------
__global__ void zero_h1_border_kernel(_Float16* __restrict__ h1, int H, int W) {
    int Wp = W + 4, Hp = H + 4;
    int idx = blockIdx.x * 256 + threadIdx.x;
    if (idx >= Hp * Wp) return;
    int y = idx / Wp, x = idx - y * Wp;
    if (y >= 2 && y < H + 2 && x >= 2 && x < W + 2) return;
    uint4 z = {0u, 0u, 0u, 0u};
    uint4* p = (uint4*)(h1 + (size_t)idx * 64);
#pragma unroll
    for (int j = 0; j < 8; ++j) p[j] = z;
}

// ---------------------------------------------------------------------------
// conv1: 1->64, 9x9, ReLU.  im2col GEMM: M=64 (4 tiles), K=96 (3 steps), N=64 px.
// Workgroup = 4 waves, one row x 64-pixel strip; wave w owns channels 16w..16w+15
// AND builds the shared B fragments for n-tile w (no redundant gathers).
// LDS tile padded to 12 rows so taps 81..95 read zeros -> branchless gather with
// compile-time offsets (one v_cndmask per element for the lane-half select).
// Output h1 channel-last f16 with 2-px halo: h1[((y+2)*(W+4)+x+2)*64 + c].
// ---------------------------------------------------------------------------
__global__ __launch_bounds__(128) void conv1_kernel(
    const float* __restrict__ in, const _Float16* __restrict__ pA,
    const float* __restrict__ bias, _Float16* __restrict__ h1, int H, int W) {
    __shared__ _Float16 tile[12 * 72];           // rows 9..11 stay zero (tap pad)
    __shared__ _Float16 bfrag[4][3][32][16];     // [nt][ks][lane][e], 12 KB
    const int tid = threadIdx.x;
    const int wave = tid >> 5, lane = tid & 31;
    const int n = lane & 15, half = lane >> 4;
    const int x0 = blockIdx.x * 64, y = blockIdx.y;

    for (int idx = tid; idx < 12 * 72; idx += 128) {
        int r = idx / 72, c = idx - r * 72;
        int gy = y + r - 4, gx = x0 + c - 4;
        float v = (r < 9 && gy >= 0 && gy < H && gx >= 0 && gx < W)
                      ? in[(size_t)gy * W + gx] : 0.f;
        tile[idx] = (_Float16)v;
    }
    __syncthreads();

    // Build B fragments for n-tile == wave; share via LDS.
    {
        const int xl = wave * 16 + n;
#pragma unroll
        for (int ks = 0; ks < 3; ++ks) {
            v16h b;
#pragma unroll
            for (int e = 0; e < 16; ++e) {
                int t0 = ks * 32 + e;            // compile-time after unroll
                int t1 = t0 + 16;
                int off0 = (t0 / 9) * 72 + (t0 % 9);
                int off1 = (t1 / 9) * 72 + (t1 % 9);
                int off = half ? off1 : off0;    // branchless lane-half select
                b[e] = tile[off + xl];
            }
            *(v16h*)(&bfrag[wave][ks][lane][0]) = b;   // 32B aligned
        }
    }
    __syncthreads();

    // A fragments: this wave's 16 output channels, 3 k-steps.
    v16h a0 = *(const v16h*)(pA + ((size_t)(wave * 3 + 0) * 32 + lane) * 16);
    v16h a1 = *(const v16h*)(pA + ((size_t)(wave * 3 + 1) * 32 + lane) * 16);
    v16h a2 = *(const v16h*)(pA + ((size_t)(wave * 3 + 2) * 32 + lane) * 16);

    float bs[8];
#pragma unroll
    for (int j = 0; j < 8; ++j) bs[j] = bias[wave * 16 + half * 8 + j];

    const int Wp = W + 4;
    for (int nt = 0; nt < 4; ++nt) {
        v16h b0 = *(const v16h*)(&bfrag[nt][0][lane][0]);
        v16h b1 = *(const v16h*)(&bfrag[nt][1][lane][0]);
        v16h b2 = *(const v16h*)(&bfrag[nt][2][lane][0]);
        v8f acc = {};
        acc = __builtin_amdgcn_wmma_f32_16x16x32_f16(false, a0, false, b0, (short)0, acc, false, false);
        acc = __builtin_amdgcn_wmma_f32_16x16x32_f16(false, a1, false, b1, (short)0, acc, false, false);
        acc = __builtin_amdgcn_wmma_f32_16x16x32_f16(false, a2, false, b2, (short)0, acc, false, false);
        int px = x0 + nt * 16 + n;
        size_t base = ((size_t)(y + 2) * Wp + (px + 2)) * 64 + wave * 16 + half * 8;
        union { _Float16 h[8]; uint4 u; } pk;
#pragma unroll
        for (int j = 0; j < 8; ++j) {
            float o = acc[j] + bs[j];
            pk.h[j] = (_Float16)(o > 0.f ? o : 0.f);
        }
        *(uint4*)(h1 + base) = pk.u;   // 16B aligned
    }
}

// ---------------------------------------------------------------------------
// conv2: 64->32, 5x5, ReLU. Implicit GEMM: M=32 (2 tiles), K=1600, N=pixels.
// Wave w: mt = w>>1, n-tile pair = w&1 -> 2 accumulators, 100 WMMAs per wave.
// h1 halo removes all bounds logic; per tap: 1 addr calc, 4 B loads with
// immediate offsets, 2 A loads (pointer bump), 4 WMMAs. h1 reads hit L2.
// Output h2 channel-last f16 (unpadded): h2[(y*W+x)*32 + c].
// ---------------------------------------------------------------------------
__global__ __launch_bounds__(128) void conv2_kernel(
    const _Float16* __restrict__ h1, const _Float16* __restrict__ pA,
    const float* __restrict__ bias, _Float16* __restrict__ h2, int H, int W) {
    const int tid = threadIdx.x;
    const int wave = tid >> 5, lane = tid & 31;
    const int mt = wave >> 1, ntp = wave & 1;
    const int x0 = blockIdx.x * 64, y = blockIdx.y;
    const int n = lane & 15, half = lane >> 4;
    const int Wp = W + 4;
    v8f acc0 = {}, acc1 = {};

    // A stream pointer for this wave's M-tile (1024 halfs = 2 k-steps per tap).
    const _Float16* pAw = pA + ((size_t)mt * 50 * 32 + lane) * 16;
    // B base column for n-tile pair (t=1 adds 16 px = 1024 halfs).
    const int xb = x0 + ntp * 32 + n + 2;

    for (int dy = -2; dy <= 2; ++dy) {
        const _Float16* rowp = h1 + ((size_t)(y + dy + 2) * Wp + xb) * 64 + half * 16;
#pragma unroll
        for (int dx = -2; dx <= 2; ++dx) {
            v16h a0 = *(const v16h*)(pAw);
            v16h a1 = *(const v16h*)(pAw + 512);
            pAw += 1024;
            const _Float16* bp = rowp + (size_t)dx * 64;
            v16h b00 = *(const v16h*)(bp);          // t=0, chunk0 (ch 0..31)
            v16h b01 = *(const v16h*)(bp + 32);     // t=0, chunk1 (ch 32..63)
            v16h b10 = *(const v16h*)(bp + 1024);   // t=1, chunk0
            v16h b11 = *(const v16h*)(bp + 1056);   // t=1, chunk1
            acc0 = __builtin_amdgcn_wmma_f32_16x16x32_f16(false, a0, false, b00, (short)0, acc0, false, false);
            acc0 = __builtin_amdgcn_wmma_f32_16x16x32_f16(false, a1, false, b01, (short)0, acc0, false, false);
            acc1 = __builtin_amdgcn_wmma_f32_16x16x32_f16(false, a0, false, b10, (short)0, acc1, false, false);
            acc1 = __builtin_amdgcn_wmma_f32_16x16x32_f16(false, a1, false, b11, (short)0, acc1, false, false);
        }
    }

    float bs[8];
#pragma unroll
    for (int j = 0; j < 8; ++j) bs[j] = bias[mt * 16 + half * 8 + j];
#pragma unroll
    for (int t = 0; t < 2; ++t) {
        v8f acc = t ? acc1 : acc0;
        int px = x0 + (ntp * 2 + t) * 16 + n;
        size_t base = ((size_t)y * W + px) * 32 + mt * 16 + half * 8;
        union { _Float16 h[8]; uint4 u; } pk;
#pragma unroll
        for (int j = 0; j < 8; ++j) {
            float o = acc[j] + bs[j];
            pk.h[j] = (_Float16)(o > 0.f ? o : 0.f);
        }
        *(uint4*)(h2 + base) = pk.u;
    }
}

// ---------------------------------------------------------------------------
// conv3: 32->1, 5x5 (1.4% of FLOPs) — plain VALU, f32 accumulate.
// ---------------------------------------------------------------------------
__global__ void conv3_kernel(const _Float16* __restrict__ h2, const float* __restrict__ w3,
                             const float* __restrict__ b3, float* __restrict__ dst, int H, int W) {
    int idx = blockIdx.x * 256 + threadIdx.x;
    if (idx >= H * W) return;
    int y = idx / W, x = idx - y * W;
    float acc = b3[0];
    for (int dy = -2; dy <= 2; ++dy) {
        int py = y + dy;
        if (py < 0 || py >= H) continue;
        for (int dx = -2; dx <= 2; ++dx) {
            int px = x + dx;
            if (px < 0 || px >= W) continue;
            int tap = (dy + 2) * 5 + (dx + 2);
            const _Float16* p = h2 + ((size_t)py * W + px) * 32;
#pragma unroll
            for (int c = 0; c < 32; ++c)
                acc += (float)p[c] * w3[c * 25 + tap];   // W3: (1,32,5,5)
        }
    }
    dst[idx] = acc;
}

// ---------------------------------------------------------------------------
// Pixel-shuffle interleave: out[2y,2x]=p0, [2y,2x+1]=p1, [2y+1,2x]=p2, [2y+1,2x+1]=p3
// ---------------------------------------------------------------------------
__global__ void interleave_kernel(const float* __restrict__ p0, const float* __restrict__ p1,
                                  const float* __restrict__ p2, const float* __restrict__ p3,
                                  float* __restrict__ out, int H, int W) {
    int idx = blockIdx.x * 256 + threadIdx.x;
    if (idx >= H * W) return;
    int y = idx / W, x = idx - y * W;
    int W2 = 2 * W;
    out[(size_t)(2 * y) * W2 + 2 * x]         = p0[idx];
    out[(size_t)(2 * y) * W2 + 2 * x + 1]     = p1[idx];
    out[(size_t)(2 * y + 1) * W2 + 2 * x]     = p2[idx];
    out[(size_t)(2 * y + 1) * W2 + 2 * x + 1] = p3[idx];
}

// ---------------------------------------------------------------------------
// MSF input: bilinear up4(out_x2) + up2(out_x4) + out_x8 (half-pixel centers).
// ---------------------------------------------------------------------------
__device__ inline float bilin_sample(const float* src, int Hs, int Ws, int y, int x, float inv_s) {
    float sy = (y + 0.5f) * inv_s - 0.5f;
    float sx = (x + 0.5f) * inv_s - 0.5f;
    float fy = floorf(sy), fx = floorf(sx);
    float wy = sy - fy, wx = sx - fx;
    int y0 = min(max((int)fy, 0), Hs - 1);
    int y1 = min(max((int)fy + 1, 0), Hs - 1);
    int x0 = min(max((int)fx, 0), Ws - 1);
    int x1 = min(max((int)fx + 1, 0), Ws - 1);
    float v00 = src[(size_t)y0 * Ws + x0], v01 = src[(size_t)y0 * Ws + x1];
    float v10 = src[(size_t)y1 * Ws + x0], v11 = src[(size_t)y1 * Ws + x1];
    return v00 * (1.f - wy) * (1.f - wx) + v01 * (1.f - wy) * wx +
           v10 * wy * (1.f - wx) + v11 * wy * wx;
}

__global__ void msf_combine_kernel(const float* __restrict__ x2, const float* __restrict__ x4,
                                   const float* __restrict__ x8v, float* __restrict__ dst) {
    int idx = blockIdx.x * 256 + threadIdx.x;
    if (idx >= 512 * 512) return;
    int y = idx >> 9, x = idx & 511;
    dst[idx] = bilin_sample(x2, 128, 128, y, x, 0.25f) +
               bilin_sample(x4, 256, 256, y, x, 0.5f) + x8v[idx];
}

// ---------------------------------------------------------------------------
// Host-side orchestration.
// ---------------------------------------------------------------------------
extern "C" void kernel_launch(void* const* d_in, const int* in_sizes, int n_in,
                              void* d_out, int out_size, void* d_ws, size_t ws_size,
                              hipStream_t stream) {
    const float* image = (const float*)d_in[0];  // (4,1,64,64)
    const float* W1    = (const float*)d_in[1];  // (13,64,1,9,9)
    const float* B1    = (const float*)d_in[2];  // (13,64)
    const float* W2    = (const float*)d_in[3];  // (13,32,64,5,5)
    const float* B2    = (const float*)d_in[4];  // (13,32)
    const float* W3    = (const float*)d_in[5];  // (13,1,32,5,5)
    const float* B3    = (const float*)d_in[6];  // (13,1)
    float* out = (float*)d_out;

    // Workspace carve-out (~55 MB total).
    char* ws = (char*)d_ws;
    size_t off = 0;
    auto carve = [&](size_t bytes) -> char* {
        char* p = ws + off;
        off += (bytes + 255) & ~(size_t)255;
        return p;
    };
    _Float16* pW1   = (_Float16*)carve((size_t)13 * 4 * 3 * 32 * 16 * sizeof(_Float16));   // 160 KB
    _Float16* pW2   = (_Float16*)carve((size_t)13 * 2 * 50 * 32 * 16 * sizeof(_Float16));  // 1.33 MB
    _Float16* h1    = (_Float16*)carve((size_t)64 * 516 * 516 * sizeof(_Float16));         // 34.1 MB (padded)
    _Float16* h2    = (_Float16*)carve((size_t)32 * 512 * 512 * sizeof(_Float16));         // 16.8 MB
    float*    pbuf  = (float*)carve((size_t)4 * 256 * 256 * sizeof(float));                // 1 MB
    float*    msfin = (float*)carve((size_t)512 * 512 * sizeof(float));                    // 1 MB
    (void)ws_size; (void)in_sizes; (void)n_in; (void)out_size;

    // Pack weights into WMMA fragment order (once per call; deterministic).
    {
        int total = 13 * 4 * 3 * 32 * 16;
        pack_w1_kernel<<<(total + 255) / 256, 256, 0, stream>>>(W1, pW1, total);
    }
    {
        int total = 13 * 2 * 50 * 32 * 16;
        pack_w2_kernel<<<(total + 255) / 256, 256, 0, stream>>>(W2, pW2, total);
    }

    float* out_x2  = out;                              // 4*128*128
    float* out_x4  = out_x2 + (size_t)4 * 128 * 128;   // 4*256*256
    float* out_x8  = out_x4 + (size_t)4 * 256 * 256;   // 4*512*512
    float* out_msf = out_x8 + (size_t)4 * 512 * 512;   // 4*512*512

    auto run_unit = [&](const float* in_img, int H, int W, int u, float* dst) {
        conv1_kernel<<<dim3(W / 64, H), 128, 0, stream>>>(
            in_img, pW1 + (size_t)u * 4 * 3 * 32 * 16, B1 + (size_t)u * 64, h1, H, W);
        conv2_kernel<<<dim3(W / 64, H), 128, 0, stream>>>(
            h1, pW2 + (size_t)u * 2 * 50 * 32 * 16, B2 + (size_t)u * 32, h2, H, W);
        conv3_kernel<<<(H * W + 255) / 256, 256, 0, stream>>>(
            h2, W3 + (size_t)u * 32 * 25, B3 + u, dst, H, W);
    };

    auto run_stage = [&](const float* in_base, int H, int W, int uoff, float* out_region) {
        int npad = (H + 4) * (W + 4);
        zero_h1_border_kernel<<<(npad + 255) / 256, 256, 0, stream>>>(h1, H, W);
        for (int b = 0; b < 4; ++b) {
            const float* in_img = in_base + (size_t)b * H * W;
            for (int i = 0; i < 4; ++i)
                run_unit(in_img, H, W, uoff + i, pbuf + (size_t)i * H * W);
            interleave_kernel<<<(H * W + 255) / 256, 256, 0, stream>>>(
                pbuf, pbuf + (size_t)H * W, pbuf + (size_t)2 * H * W, pbuf + (size_t)3 * H * W,
                out_region + (size_t)b * 4 * H * W, H, W);
        }
    };

    run_stage(image,  64,  64,  0, out_x2);
    run_stage(out_x2, 128, 128, 4, out_x4);
    run_stage(out_x4, 256, 256, 8, out_x8);

    {
        int npad = 516 * 516;
        zero_h1_border_kernel<<<(npad + 255) / 256, 256, 0, stream>>>(h1, 512, 512);
        for (int b = 0; b < 4; ++b) {
            msf_combine_kernel<<<(512 * 512 + 255) / 256, 256, 0, stream>>>(
                out_x2 + (size_t)b * 128 * 128, out_x4 + (size_t)b * 256 * 256,
                out_x8 + (size_t)b * 512 * 512, msfin);
            run_unit(msfin, 512, 512, 12, out_msf + (size_t)b * 512 * 512);
        }
    }
}
